// U_EM_DIRICHLET_EMBT_68874095559200
// MI455X (gfx1250) — matrix operand order repaired and checked by reading, not compiled, not run on hardware
//
#include <hip/hip_runtime.h>

// ---------------------------------------------------------------------------
// EM-Dirichlet forward for gfx1250 (MI455X).
//  T=500 tasks, S=320 support, Q=1024 query, C=F=64 classes.
//  GEMMs via v_wmma_f32_16x16x32_bf16 (bf16 operands, f32 accumulate).
//  Alpha MM loop: 1 wave32 per (t,c) row, 2 elems/lane, all in registers.
// ---------------------------------------------------------------------------

typedef __attribute__((ext_vector_type(16))) __bf16 v16bf;
typedef __attribute__((ext_vector_type(8)))  __bf16 bf16x8;
typedef __attribute__((ext_vector_type(8)))  float  v8f;

#define EPSF      1e-15f
#define ZERO_CURV 1.6449340668482264f   // pi^2/6

__device__ __forceinline__ float wave_sum(float v) {
#pragma unroll
  for (int o = 16; o > 0; o >>= 1) v += __shfl_xor(v, o, 32);
  return v;
}
__device__ __forceinline__ float wave_max(float v) {
#pragma unroll
  for (int o = 16; o > 0; o >>= 1) v = fmaxf(v, __shfl_xor(v, o, 32));
  return v;
}
// float -> bf16 (RNE) as raw u16 so host code never touches __bf16
__device__ __forceinline__ unsigned short f2bf(float x) {
  unsigned u = __builtin_bit_cast(unsigned, x);
  u += 0x7fffu + ((u >> 16) & 1u);
  return (unsigned short)(u >> 16);
}
__device__ __forceinline__ float softplusf(float x) {
  return (x > 20.f) ? x : log1pf(expf(x));
}
// digamma for x>0: shift-up recurrence + asymptotic series
__device__ __forceinline__ float digammaf_pos(float x) {
  float r = 0.f;
#pragma unroll 1
  for (int i = 0; i < 8 && x < 6.f; ++i) { r -= 1.f / x; x += 1.f; }
  float inv = 1.f / x, inv2 = inv * inv;
  r += logf(x) - 0.5f * inv
     - inv2 * (0.08333333333f - inv2 * (0.008333333333f - inv2 * 0.003968253968f));
  return r;
}

// ---------------------------------------------------------------------------
// K1: softmax query rows -> U0 (f32, d_out), logXq (bf16 row-major),
//     logXq^T and U0^T (bf16 transposed) for WMMA operands.
//     1 wave per (t,q) row of 64; lane holds elems {l, l+32}.
// ---------------------------------------------------------------------------
__global__ void k_prep_query(const float* __restrict__ Xq,
                             const float* __restrict__ embt_p,
                             float* __restrict__ U0,
                             unsigned short* __restrict__ Lrm,
                             unsigned short* __restrict__ Lt,
                             unsigned short* __restrict__ Ut) {
  const int wave = threadIdx.x >> 5, lane = threadIdx.x & 31;
  const long long row = (long long)blockIdx.x * 8 + wave;  // t*1024 + q
  const float embt = embt_p[0];
  const float* x = Xq + row * 64;
  float x0 = embt * x[lane], x1 = embt * x[lane + 32];
  float m = wave_max(fmaxf(x0, x1));
  float e0 = expf(x0 - m), e1 = expf(x1 - m);
  float inv = 1.f / wave_sum(e0 + e1);
  float p0 = e0 * inv, p1 = e1 * inv;
  float* u = U0 + row * 64;
  u[lane] = p0; u[lane + 32] = p1;
  float l0 = logf(p0 + EPSF), l1 = logf(p1 + EPSF);
  Lrm[row * 64 + lane] = f2bf(l0);
  Lrm[row * 64 + lane + 32] = f2bf(l1);
  const long long t = row >> 10, q = row & 1023;
  unsigned short* lt = Lt + t * (64ll * 1024);
  unsigned short* ut = Ut + t * (64ll * 1024);
  lt[(long long)lane * 1024 + q]        = f2bf(l0);
  lt[(long long)(lane + 32) * 1024 + q] = f2bf(l1);
  ut[(long long)lane * 1024 + q]        = f2bf(p0);
  ut[(long long)(lane + 32) * 1024 + q] = f2bf(p1);
}

// ---------------------------------------------------------------------------
// K2: support stats. One block per task; sup_stat tile lives in LDS,
//     scatter-add by class with LDS float atomics.
// ---------------------------------------------------------------------------
__global__ void k_prep_support(const float* __restrict__ Xs,
                               const int* __restrict__ ysup,
                               const float* __restrict__ embt_p,
                               float* __restrict__ sup_stat,
                               float* __restrict__ y_s_sum) {
  __shared__ float sup[64 * 64];
  __shared__ float cnt[64];
  const int t = blockIdx.x;
  for (int i = threadIdx.x; i < 64 * 64; i += blockDim.x) sup[i] = 0.f;
  if (threadIdx.x < 64) cnt[threadIdx.x] = 0.f;
  __syncthreads();
  const int wave = threadIdx.x >> 5, lane = threadIdx.x & 31;
  const float embt = embt_p[0];
  for (int s = wave; s < 320; s += 8) {
    const float* x = Xs + ((long long)t * 320 + s) * 64;
    float x0 = embt * x[lane], x1 = embt * x[lane + 32];
    float m = wave_max(fmaxf(x0, x1));
    float e0 = expf(x0 - m), e1 = expf(x1 - m);
    float inv = 1.f / wave_sum(e0 + e1);
    float l0 = logf(e0 * inv + EPSF), l1 = logf(e1 * inv + EPSF);
    int y = ysup[t * 320 + s];
    atomicAdd(&sup[y * 64 + lane], l0);
    atomicAdd(&sup[y * 64 + lane + 32], l1);
    if (lane == 0) atomicAdd(&cnt[y], 1.f);
  }
  __syncthreads();
  for (int i = threadIdx.x; i < 64 * 64; i += blockDim.x)
    sup_stat[(long long)t * 4096 + i] = sup[i];
  if (threadIdx.x < 64) y_s_sum[t * 64 + threadIdx.x] = cnt[threadIdx.x];
}

// ---------------------------------------------------------------------------
// K3: u_sum[t,c] = sum_q U[t,q,c]  (coalesced column sums)
// ---------------------------------------------------------------------------
__global__ void k_usum(const float* __restrict__ U, float* __restrict__ u_sum) {
  __shared__ float part[4][64];
  const int t = blockIdx.x;
  const int c = threadIdx.x & 63, p = threadIdx.x >> 6;
  const float* u = U + (long long)t * 1024 * 64;
  float s = 0.f;
  for (int q = p; q < 1024; q += 4) s += u[(long long)q * 64 + c];
  part[p][c] = s;
  __syncthreads();
  if (threadIdx.x < 64)
    u_sum[t * 64 + threadIdx.x] = part[0][threadIdx.x] + part[1][threadIdx.x] +
                                  part[2][threadIdx.x] + part[3][threadIdx.x];
}

// ---------------------------------------------------------------------------
// K4: y_cst = (sup_stat + U^T @ logXq) / (y_s_sum + u_sum)
//     One block (4 waves) per task; wave w owns c-rows [16w,16w+16),
//     4 WMMA n-tiles, K=1024 in steps of 32 (bf16).
//     A = U^T (from Ut), B = logXq (from Lt, transposed storage).
// ---------------------------------------------------------------------------
__global__ void k_gemm1_ycst(const unsigned short* __restrict__ Ut,
                             const unsigned short* __restrict__ Lt,
                             const float* __restrict__ sup_stat,
                             const float* __restrict__ y_s_sum,
                             const float* __restrict__ u_sum,
                             float* __restrict__ y_cst) {
  const int t = blockIdx.x;
  const int wave = threadIdx.x >> 5, lane = threadIdx.x & 31;
  const int lhi = lane >> 4, llo = lane & 15;
  const unsigned short* ut = Ut + (long long)t * 64 * 1024;
  const unsigned short* lt = Lt + (long long)t * 64 * 1024;
  v8f acc[4];
#pragma unroll
  for (int n = 0; n < 4; ++n) acc[n] = (v8f){0.f,0.f,0.f,0.f,0.f,0.f,0.f,0.f};
  const long long arow = wave * 16 + llo;            // c row for A
  const int koffA = lhi * 16, koffB = lhi * 16;
  for (int kq = 0; kq < 1024; kq += 32) {
    union { v16bf v; bf16x8 h[2]; } a;
    a.h[0] = *(const bf16x8*)(ut + arow * 1024 + kq + koffA);
    a.h[1] = *(const bf16x8*)(ut + arow * 1024 + kq + 8 + koffA);
#pragma unroll
    for (int n = 0; n < 4; ++n) {
      v16bf b = *(const v16bf*)(lt + (long long)(n * 16 + llo) * 1024 + kq + koffB);
      acc[n] = __builtin_amdgcn_wmma_f32_16x16x32_bf16(
          false, a.v, false, b, (short)0, acc[n], false, false);
    }
  }
  const float* sup = sup_stat + (long long)t * 4096;
  const float* ys = y_s_sum + t * 64;
  const float* us = u_sum + t * 64;
#pragma unroll
  for (int n = 0; n < 4; ++n) {
#pragma unroll
    for (int r = 0; r < 8; ++r) {
      int ci = wave * 16 + r + lhi * 8;   // D: VGPR r -> M=r (lanes<16) / r+8
      int fi = n * 16 + llo;              //    lane  -> N
      float denom = ys[ci] + us[ci];
      y_cst[(long long)t * 4096 + ci * 64 + fi] =
          (sup[ci * 64 + fi] + acc[n][r]) / denom;
    }
  }
}

// ---------------------------------------------------------------------------
// K5: 50 MM fixed-point iterations on alpha, fully in registers.
//     1 wave per (t,c) row of 64; lane holds {f=l, f=l+32}.
//     Epilogue: alpha out, (alpha-1) as bf16 [c][f] for GEMM2 B operand,
//     rowconst = lgamma(sum a) - sum lgamma(a) + softplus(gamma)*V.
// ---------------------------------------------------------------------------
__device__ __forceinline__ float mm_step(float a, float y, float dgs) {
  float dg = digammaf_pos(a + 1.f);
  float lg = lgammaf(a + 1.f);
  float curv = (a > 1e-11f) ? fabsf(2.f * (-lg + dg * a) / (a * a)) : ZERO_CURV;
  float b = dg - dgs - curv * a - y;
  return (-b + sqrtf(b * b + 4.f * curv)) / (2.f * curv);
}

__global__ void k_alpha(const float* __restrict__ y_cst,
                        float* __restrict__ alpha,
                        const float* __restrict__ u_sum,
                        const float* __restrict__ gamma_p,
                        float* __restrict__ rowconst,
                        unsigned short* __restrict__ Am1,
                        int first) {
  const int wave = threadIdx.x >> 5, lane = threadIdx.x & 31;
  const long long row = (long long)blockIdx.x * 8 + wave;  // t*64 + c
  const float* y = y_cst + row * 64;
  const float y0 = y[lane], y1 = y[lane + 32];
  float a0, a1;
  if (first) { a0 = 1.f; a1 = 1.f; }
  else { a0 = alpha[row * 64 + lane]; a1 = alpha[row * 64 + lane + 32]; }
#pragma unroll 1
  for (int it = 0; it < 50; ++it) {
    float dgs = digammaf_pos(wave_sum(a0 + a1));
    a0 = mm_step(a0, y0, dgs);
    a1 = mm_step(a1, y1, dgs);
  }
  alpha[row * 64 + lane] = a0;
  alpha[row * 64 + lane + 32] = a1;
  float asum = wave_sum(a0 + a1);
  float slg = wave_sum(lgammaf(a0) + lgammaf(a1));
  float rowA = lgammaf(asum) - slg;
  if (lane == 0) {
    float g = softplusf(gamma_p[0]);
    float V = logf(u_sum[row] * (1.f / 1024.f) + EPSF) + 1.f;
    rowconst[row] = rowA + g * V;
  }
  Am1[row * 64 + lane] = f2bf(a0 - 1.f);
  Am1[row * 64 + lane + 32] = f2bf(a1 - 1.f);
}

// ---------------------------------------------------------------------------
// K6: logits = logXq @ (alpha-1)^T + rowconst, then U = softmax(inv_t*logits).
//     One block (4 waves) per 16 q-rows; wave w owns c-tile w; K=64 (2 steps).
//     Full 64-class strip lands in LDS -> fused row softmax.
//     Writes U (f32, d_out) and U^T (bf16) for next layer's GEMM1.
// ---------------------------------------------------------------------------
__global__ void k_gemm2_softmax(const unsigned short* __restrict__ Lrm,
                                const unsigned short* __restrict__ Am1,
                                const float* __restrict__ rowconst,
                                const float* __restrict__ t_p,
                                float* __restrict__ U,
                                unsigned short* __restrict__ Ut) {
  __shared__ float tile[16][64];
  __shared__ float rmax[16], rinv[16];
  const int t = blockIdx.y;
  const int qBase = blockIdx.x * 16;
  const int wave = threadIdx.x >> 5, lane = threadIdx.x & 31;
  const int lhi = lane >> 4, llo = lane & 15;
  const unsigned short* L = Lrm + ((long long)t * 1024 + qBase) * 64;
  const unsigned short* B = Am1 + (long long)t * 4096;
  v8f acc = (v8f){0.f,0.f,0.f,0.f,0.f,0.f,0.f,0.f};
#pragma unroll
  for (int kf = 0; kf < 64; kf += 32) {
    union { v16bf v; bf16x8 h[2]; } a;
    a.h[0] = *(const bf16x8*)(L + (long long)llo * 64 + kf + lhi * 16);
    a.h[1] = *(const bf16x8*)(L + (long long)llo * 64 + kf + 8 + lhi * 16);
    v16bf b = *(const v16bf*)(B + (long long)(wave * 16 + llo) * 64 + kf + lhi * 16);
    acc = __builtin_amdgcn_wmma_f32_16x16x32_bf16(
        false, a.v, false, b, (short)0, acc, false, false);
  }
  const float inv_t = 1.f / (softplusf(t_p[0]) + 1.f);
  const float* rc = rowconst + t * 64;
#pragma unroll
  for (int r = 0; r < 8; ++r) {
    int qr = r + lhi * 8;          // D: M = local q row
    int ci = wave * 16 + llo;      //    N = class
    tile[qr][ci] = inv_t * (acc[r] + rc[ci]);
  }
  __syncthreads();
  if (threadIdx.x < 16) {
    int r = threadIdx.x;
    float m = -1e30f;
    for (int c = 0; c < 64; ++c) m = fmaxf(m, tile[r][c]);
    float s = 0.f;
    for (int c = 0; c < 64; ++c) s += expf(tile[r][c] - m);
    rmax[r] = m; rinv[r] = 1.f / s;
  }
  __syncthreads();
  float* uo = U + ((long long)t * 1024 + qBase) * 64;
  unsigned short* ut = Ut + (long long)t * 64 * 1024;
  for (int i = threadIdx.x; i < 1024; i += 128) {
    int r = i >> 6, c = i & 63;
    float p = expf(tile[r][c] - rmax[r]) * rinv[r];
    uo[(long long)r * 64 + c] = p;
    ut[(long long)c * 1024 + qBase + r] = f2bf(p);
  }
}

// ---------------------------------------------------------------------------
extern "C" void kernel_launch(void* const* d_in, const int* in_sizes, int n_in,
                              void* d_out, int out_size, void* d_ws, size_t ws_size,
                              hipStream_t stream) {
  const float* Xs    = (const float*)d_in[0];
  const float* Xq    = (const float*)d_in[1];
  const int*   ysup  = (const int*)d_in[2];
  const float* embt  = (const float*)d_in[3];
  const float* gamma = (const float*)d_in[4];
  const float* tt    = (const float*)d_in[5];
  (void)in_sizes; (void)n_in; (void)out_size; (void)ws_size;

  const int T = 500, Q = 1024, C = 64;
  char* ws = (char*)d_ws;
  size_t off = 0;
  auto carve = [&](size_t bytes) -> char* {
    char* p = ws + off;
    off += (bytes + 255) & ~(size_t)255;
    return p;
  };
  unsigned short* Lt  = (unsigned short*)carve((size_t)T * C * Q * 2);  // logXq^T
  unsigned short* Lrm = (unsigned short*)carve((size_t)T * Q * C * 2);  // logXq
  unsigned short* Ut  = (unsigned short*)carve((size_t)T * C * Q * 2);  // U^T
  unsigned short* Am1 = (unsigned short*)carve((size_t)T * C * C * 2);  // alpha-1
  float* sup   = (float*)carve((size_t)T * C * C * 4);
  float* yss   = (float*)carve((size_t)T * C * 4);
  float* usum  = (float*)carve((size_t)T * C * 4);
  float* alpha = (float*)carve((size_t)T * C * C * 4);
  float* ycst  = (float*)carve((size_t)T * C * C * 4);
  float* rowc  = (float*)carve((size_t)T * C * 4);
  float* U = (float*)d_out;  // U lives in d_out (U0 = softmax(query))

  k_prep_query<<<T * Q / 8, 256, 0, stream>>>(Xq, embt, U, Lrm, Lt, Ut);
  k_prep_support<<<T, 256, 0, stream>>>(Xs, ysup, embt, sup, yss);
  for (int layer = 0; layer < 3; ++layer) {
    k_usum<<<T, 256, 0, stream>>>(U, usum);
    k_gemm1_ycst<<<T, 128, 0, stream>>>(Ut, Lt, sup, yss, usum, ycst);
    k_alpha<<<T * C / 8, 256, 0, stream>>>(ycst, alpha, usum, gamma, rowc, Am1,
                                           layer == 0 ? 1 : 0);
    k_gemm2_softmax<<<dim3(Q / 16, T), 128, 0, stream>>>(Lrm, Am1, rowc, tt, U, Ut);
  }
}
